// InformerProbSparseAttention_66460323938856
// MI455X (gfx1250) — matrix operand order repaired and checked by reading, not compiled, not run on hardware
//
#include <hip/hip_runtime.h>
#include <hip/hip_bf16.h>
#include <float.h>

typedef __attribute__((ext_vector_type(16))) __bf16 v16bf;
typedef __attribute__((ext_vector_type(8)))  float  v8f;

#define EMBED 512
#define T_LEN 2048
#define BATCH 4
#define NHEAD 8
#define HDIM  64
#define BH    32          // BATCH*NHEAD
#define U_TOP 40          // min(5*ceil(log1p(2048)), 2048)
#define U_PAD 48          // padded to 3x16 row tiles

// ---------- WMMA fragment helpers (contiguous k order, wave32) ----------

// 16-bit A 16x32: lane m = lane&15; half = lane>>4.
// v16bf elems 0..15 = K[half*16 .. half*16+15]
__device__ __forceinline__ v16bf load_a_frag(const __bf16* row, int half) {
    v16bf a;
    __builtin_memcpy(&a, row + half * 16, 32);
    return a;
}

// 16-bit B 32x16: lane n = lane&15; elems i = K[half*16 + i] of column n.
// colsrc points at the contiguous K-run for this lane's column.
__device__ __forceinline__ v16bf load_b_frag(const __bf16* colsrc, int half) {
    v16bf b;
    __builtin_memcpy(&b, colsrc + half * 16, 32);
    return b;
}

__device__ __forceinline__ v8f wmma_bf16(v16bf a, v16bf b, v8f c) {
    return __builtin_amdgcn_wmma_f32_16x16x32_bf16(
        /*neg_a=*/false, a, /*neg_b=*/false, b,
        /*c_mod=*/(short)0, c, /*reuse_a=*/false, /*reuse_b=*/false);
}

// ---------- kernels ----------

__global__ void cvt_f32_bf16(const float* __restrict__ src, __bf16* __restrict__ dst, int n) {
    int i = blockIdx.x * blockDim.x + threadIdx.x;
    if (i < n) dst[i] = (__bf16)src[i];
}

// out = X(8192x512,bf16) @ W(512x512,bf16 row-major)^T + bias, one wave per 16x16 tile.
// mode 0: Q  -> bf16 [bh][t][d], scaled by 1/8
// mode 1: K  -> bf16 [bh][t][d]
// mode 2: V  -> bf16 [bh][d][t] (transposed for P@V B-fragments)
// mode 3: O  -> f32  row-major [m][n] (final output)
__global__ void gemm_proj(const __bf16* __restrict__ X, const __bf16* __restrict__ W,
                          const float* __restrict__ bias, void* __restrict__ outp,
                          int mode, float scale) {
    const int wid  = (blockIdx.x * blockDim.x + threadIdx.x) >> 5;
    const int lane = threadIdx.x & 31;
    const int NT   = EMBED / 16;                       // 32 col tiles
    const int MT   = (BATCH * T_LEN) / 16;             // 512 row tiles
    const int mt = wid / NT, nt = wid % NT;
    if (mt >= MT) return;
    const int m0 = mt * 16, n0 = nt * 16;
    const int ln = lane & 15, half = lane >> 4;

    const __bf16* arow = X + (size_t)(m0 + ln) * EMBED;
    const __bf16* brow = W + (size_t)(n0 + ln) * EMBED;   // column n of B = row n of W
    v8f c = {};
#pragma unroll 4
    for (int k0 = 0; k0 < EMBED; k0 += 32)
        c = wmma_bf16(load_a_frag(arow + k0, half), load_b_frag(brow + k0, half), c);

    const int col = n0 + ln;
    const float bcol = bias[col];
#pragma unroll
    for (int r = 0; r < 8; ++r) {
        const int row = m0 + r + 8 * half;
        const float v = (c[r] + bcol) * scale;
        if (mode == 3) {
            ((float*)outp)[(size_t)row * EMBED + col] = v;
        } else {
            const int b = row >> 11, t = row & (T_LEN - 1);
            const int h = col >> 6,  d = col & (HDIM - 1);
            __bf16* o = (__bf16*)outp;
            if (mode == 2)
                o[((size_t)(b * NHEAD + h) * HDIM + d) * T_LEN + t] = (__bf16)v;
            else
                o[((size_t)(b * NHEAD + h) * T_LEN + t) * HDIM + d] = (__bf16)v;
        }
    }
}

// Fused sampled QK^T + sparsity = rowmax - rowsum/L_k.  One wave per (bh, 16-row q tile).
__global__ void qk_sparsity(const __bf16* __restrict__ q, const __bf16* __restrict__ k,
                            const int* __restrict__ samp, float* __restrict__ sparsity) {
    const int wid  = (blockIdx.x * blockDim.x + threadIdx.x) >> 5;
    const int lane = threadIdx.x & 31;
    const int bh = wid >> 7, rt = wid & 127;
    if (bh >= BH) return;
    const int ln = lane & 15, half = lane >> 4;
    const __bf16* qrow = q + ((size_t)bh * T_LEN + rt * 16 + ln) * HDIM;

    float rmax[8];
    v8f   csum = {};
#pragma unroll
    for (int r = 0; r < 8; ++r) rmax[r] = -FLT_MAX;

    for (int c0 = 0; c0 < T_LEN; c0 += 16) {
        const int idx = samp[c0 + ln];
        const __bf16* krow = k + ((size_t)bh * T_LEN + idx) * HDIM;
        v8f c = {};
        c = wmma_bf16(load_a_frag(qrow,      half), load_b_frag(krow,      half), c);
        c = wmma_bf16(load_a_frag(qrow + 32, half), load_b_frag(krow + 32, half), c);
#pragma unroll
        for (int r = 0; r < 8; ++r) {
            float x = c[r];
            csum[r] += x;                        // elementwise; collapsed once at the end
            x = fmaxf(x, __shfl_xor(x, 1, 16));
            x = fmaxf(x, __shfl_xor(x, 2, 16));
            x = fmaxf(x, __shfl_xor(x, 4, 16));
            x = fmaxf(x, __shfl_xor(x, 8, 16));
            rmax[r] = fmaxf(rmax[r], x);
        }
    }
#pragma unroll
    for (int r = 0; r < 8; ++r) {
        float s = csum[r];
        s += __shfl_xor(s, 1, 16);
        s += __shfl_xor(s, 2, 16);
        s += __shfl_xor(s, 4, 16);
        s += __shfl_xor(s, 8, 16);
        if (ln == 0) {   // lanes 0 & 16 write rows r and r+8
            const int row = rt * 16 + r + 8 * half;
            sparsity[(size_t)bh * T_LEN + row] = rmax[r] - s * (1.0f / (float)T_LEN);
        }
    }
}

// Top-40 per bh via iterative argmax in LDS; pad indices to 48.
__global__ void topk_kernel(const float* __restrict__ sparsity, int* __restrict__ topidx) {
    const int bh = blockIdx.x;
    const int tid = threadIdx.x;
    __shared__ float sv[T_LEN];
    __shared__ float rbest[256];
    __shared__ int   ribest[256];
    for (int j = tid; j < T_LEN; j += 256) sv[j] = sparsity[(size_t)bh * T_LEN + j];
    __syncthreads();
    for (int it = 0; it < U_PAD; ++it) {
        if (it < U_TOP) {
            float best = -FLT_MAX; int bix = 0;
            for (int j = tid; j < T_LEN; j += 256) {
                float x = sv[j];
                if (x > best) { best = x; bix = j; }
            }
            rbest[tid] = best; ribest[tid] = bix;
            __syncthreads();
            for (int s = 128; s > 0; s >>= 1) {
                if (tid < s && rbest[tid + s] > rbest[tid]) {
                    rbest[tid] = rbest[tid + s]; ribest[tid] = ribest[tid + s];
                }
                __syncthreads();
            }
            if (tid == 0) { topidx[bh * U_PAD + it] = ribest[0]; sv[ribest[0]] = -FLT_MAX; }
            __syncthreads();
        } else if (tid == 0) {
            topidx[bh * U_PAD + it] = 0;
        }
    }
}

// S = q[top_idx] @ K^T ; one wave per (bh, row tile, 256-column block).
__global__ void attn_scores(const __bf16* __restrict__ q, const __bf16* __restrict__ k,
                            const int* __restrict__ topidx, float* __restrict__ S) {
    const int wid  = (blockIdx.x * blockDim.x + threadIdx.x) >> 5;
    const int lane = threadIdx.x & 31;
    if (wid >= BH * 3 * 8) return;
    const int bh = wid / 24, rem = wid % 24;
    const int rt = rem / 8, cb = rem % 8;
    const int ln = lane & 15, half = lane >> 4;
    const int tq = topidx[bh * U_PAD + rt * 16 + ln];
    const __bf16* qrow = q + ((size_t)bh * T_LEN + tq) * HDIM;
    for (int ct = 0; ct < 16; ++ct) {
        const int t0 = cb * 256 + ct * 16;
        const __bf16* krow = k + ((size_t)bh * T_LEN + t0 + ln) * HDIM;
        v8f c = {};
        c = wmma_bf16(load_a_frag(qrow,      half), load_b_frag(krow,      half), c);
        c = wmma_bf16(load_a_frag(qrow + 32, half), load_b_frag(krow + 32, half), c);
#pragma unroll
        for (int r = 0; r < 8; ++r)
            S[((size_t)bh * U_PAD + rt * 16 + r + 8 * half) * T_LEN + t0 + ln] = c[r];
    }
}

// Row softmax of S -> bf16 P (pad rows written as zero so P@V tiles are uniform).
__global__ void softmax_rows(const float* __restrict__ S, __bf16* __restrict__ P) {
    const int row = blockIdx.x;               // bh*U_PAD + u
    const int u   = row % U_PAD;
    const int tid = threadIdx.x;
    if (u >= U_TOP) {
        for (int j = tid; j < T_LEN; j += 256) P[(size_t)row * T_LEN + j] = (__bf16)0.0f;
        return;
    }
    __shared__ float sv[T_LEN];
    __shared__ float red[256];
    const float* srow = S + (size_t)row * T_LEN;
    float mx = -FLT_MAX;
    for (int j = tid; j < T_LEN; j += 256) { float x = srow[j]; sv[j] = x; mx = fmaxf(mx, x); }
    red[tid] = mx; __syncthreads();
    for (int s = 128; s > 0; s >>= 1) { if (tid < s) red[tid] = fmaxf(red[tid], red[tid + s]); __syncthreads(); }
    mx = red[0]; __syncthreads();
    float sum = 0.f;
    for (int j = tid; j < T_LEN; j += 256) { float e = __expf(sv[j] - mx); sv[j] = e; sum += e; }
    red[tid] = sum; __syncthreads();
    for (int s = 128; s > 0; s >>= 1) { if (tid < s) red[tid] += red[tid + s]; __syncthreads(); }
    const float inv = 1.0f / red[0];
    for (int j = tid; j < T_LEN; j += 256) P[(size_t)row * T_LEN + j] = (__bf16)(sv[j] * inv);
}

// attn_out = P @ V ; one wave per (bh, row tile, 16-wide d tile), K=2048.
__global__ void pv_gemm(const __bf16* __restrict__ P, const __bf16* __restrict__ vT,
                        float* __restrict__ AO) {
    const int wid  = (blockIdx.x * blockDim.x + threadIdx.x) >> 5;
    const int lane = threadIdx.x & 31;
    if (wid >= BH * 3 * 4) return;
    const int bh = wid / 12, rem = wid % 12;
    const int rt = rem / 4, nt = rem % 4;
    const int ln = lane & 15, half = lane >> 4;
    const __bf16* arow = P  + ((size_t)bh * U_PAD + rt * 16 + ln) * T_LEN;
    const __bf16* brow = vT + ((size_t)bh * HDIM  + nt * 16 + ln) * T_LEN;
    v8f c = {};
#pragma unroll 4
    for (int k0 = 0; k0 < T_LEN; k0 += 32)
        c = wmma_bf16(load_a_frag(arow + k0, half), load_b_frag(brow + k0, half), c);
#pragma unroll
    for (int r = 0; r < 8; ++r)
        AO[((size_t)bh * U_PAD + rt * 16 + r + 8 * half) * HDIM + nt * 16 + ln] = c[r];
}

__global__ void vmean_kernel(const __bf16* __restrict__ vT, float* __restrict__ vmean) {
    const int row = blockIdx.x;               // bh*HDIM + d
    const int tid = threadIdx.x;
    __shared__ float red[256];
    float s = 0.f;
    const __bf16* src = vT + (size_t)row * T_LEN;
    for (int j = tid; j < T_LEN; j += 256) s += (float)src[j];
    red[tid] = s; __syncthreads();
    for (int st = 128; st > 0; st >>= 1) { if (tid < st) red[tid] += red[tid + st]; __syncthreads(); }
    if (tid == 0) vmean[row] = red[0] * (1.0f / (float)T_LEN);
}

__global__ void ctx_fill(const float* __restrict__ vmean, __bf16* __restrict__ ctx) {
    const size_t i = (size_t)blockIdx.x * blockDim.x + threadIdx.x;
    if (i >= (size_t)BATCH * T_LEN * EMBED) return;
    const int e = (int)(i % EMBED);
    const int b = (int)(i / ((size_t)T_LEN * EMBED));
    const int h = e >> 6, d = e & (HDIM - 1);
    ctx[i] = (__bf16)vmean[(b * NHEAD + h) * HDIM + d];
}

__global__ void ctx_scatter(const float* __restrict__ AO, const int* __restrict__ topidx,
                            __bf16* __restrict__ ctx) {
    const int i = blockIdx.x * blockDim.x + threadIdx.x;
    if (i >= BH * U_TOP * HDIM) return;
    const int d  = i & (HDIM - 1);
    const int u  = (i >> 6) % U_TOP;
    const int bh = i / (U_TOP * HDIM);
    const int t  = topidx[bh * U_PAD + u];
    const int b = bh >> 3, h = bh & 7;
    ctx[((size_t)b * T_LEN + t) * EMBED + h * HDIM + d] =
        (__bf16)AO[((size_t)bh * U_PAD + u) * HDIM + d];
}

// ---------- launch ----------

extern "C" void kernel_launch(void* const* d_in, const int* in_sizes, int n_in,
                              void* d_out, int out_size, void* d_ws, size_t ws_size,
                              hipStream_t stream) {
    (void)in_sizes; (void)n_in; (void)out_size; (void)ws_size;
    const float* hs   = (const float*)d_in[0];
    const int*   samp = (const int*)d_in[1];
    const float* Wq = (const float*)d_in[2];
    const float* bq = (const float*)d_in[3];
    const float* Wk = (const float*)d_in[4];
    const float* bk = (const float*)d_in[5];
    const float* Wv = (const float*)d_in[6];
    const float* bv = (const float*)d_in[7];
    const float* Wo = (const float*)d_in[8];
    const float* bo = (const float*)d_in[9];

    char* ws = (char*)d_ws;
    size_t off = 0;
    auto alloc = [&](size_t bytes) -> void* {
        void* p = ws + off;
        off = (off + bytes + 255) & ~(size_t)255;
        return p;
    };
    const size_t MROWS = (size_t)BATCH * T_LEN;            // 8192
    __bf16* Xbf  = (__bf16*)alloc(MROWS * EMBED * 2);
    __bf16* Wqb  = (__bf16*)alloc((size_t)EMBED * EMBED * 2);
    __bf16* Wkb  = (__bf16*)alloc((size_t)EMBED * EMBED * 2);
    __bf16* Wvb  = (__bf16*)alloc((size_t)EMBED * EMBED * 2);
    __bf16* Wob  = (__bf16*)alloc((size_t)EMBED * EMBED * 2);
    __bf16* qbf  = (__bf16*)alloc((size_t)BH * T_LEN * HDIM * 2);
    __bf16* kbf  = (__bf16*)alloc((size_t)BH * T_LEN * HDIM * 2);
    __bf16* vT   = (__bf16*)alloc((size_t)BH * HDIM * T_LEN * 2);
    float*  spar = (float*)alloc((size_t)BH * T_LEN * 4);
    int*    tidx = (int*)alloc((size_t)BH * U_PAD * 4);
    float*  S    = (float*)alloc((size_t)BH * U_PAD * T_LEN * 4);
    __bf16* P    = (__bf16*)alloc((size_t)BH * U_PAD * T_LEN * 2);
    float*  AO   = (float*)alloc((size_t)BH * U_PAD * HDIM * 4);
    float*  vmn  = (float*)alloc((size_t)BH * HDIM * 4);
    __bf16* ctx  = (__bf16*)alloc(MROWS * EMBED * 2);

    // 1) fp32 -> bf16 conversions
    {
        int n = (int)(MROWS * EMBED);
        cvt_f32_bf16<<<(n + 255) / 256, 256, 0, stream>>>(hs, Xbf, n);
        int nw = EMBED * EMBED;
        cvt_f32_bf16<<<(nw + 255) / 256, 256, 0, stream>>>(Wq, Wqb, nw);
        cvt_f32_bf16<<<(nw + 255) / 256, 256, 0, stream>>>(Wk, Wkb, nw);
        cvt_f32_bf16<<<(nw + 255) / 256, 256, 0, stream>>>(Wv, Wvb, nw);
        cvt_f32_bf16<<<(nw + 255) / 256, 256, 0, stream>>>(Wo, Wob, nw);
    }

    // 2) Q/K/V projections (16384 tiles, 8 waves/block)
    const int projBlocks = (int)((MROWS / 16) * (EMBED / 16) / 8);   // 2048
    gemm_proj<<<projBlocks, 256, 0, stream>>>(Xbf, Wqb, bq, (void*)qbf, 0, 0.125f);
    gemm_proj<<<projBlocks, 256, 0, stream>>>(Xbf, Wkb, bk, (void*)kbf, 1, 1.0f);
    gemm_proj<<<projBlocks, 256, 0, stream>>>(Xbf, Wvb, bv, (void*)vT,  2, 1.0f);

    // 3) fused sampled QK^T + sparsity  (BH * 128 waves)
    qk_sparsity<<<(BH * 128) / 8, 256, 0, stream>>>(qbf, kbf, samp, spar);

    // 4) top-k
    topk_kernel<<<BH, 256, 0, stream>>>(spar, tidx);

    // 5) reduced attention scores, softmax, P@V
    attn_scores<<<(BH * 3 * 8) / 8, 256, 0, stream>>>(qbf, kbf, tidx, S);
    softmax_rows<<<BH * U_PAD, 256, 0, stream>>>(S, P);
    vmean_kernel<<<BH * HDIM, 256, 0, stream>>>(vT, vmn);
    pv_gemm<<<(BH * 3 * 4) / 8 + 1, 256, 0, stream>>>(P, vT, AO);

    // 6) context assembly: broadcast v-mean then scatter attn rows
    {
        int n = (int)(MROWS * EMBED);
        ctx_fill<<<(n + 255) / 256, 256, 0, stream>>>(vmn, ctx);
        int ns = BH * U_TOP * HDIM;
        ctx_scatter<<<(ns + 255) / 256, 256, 0, stream>>>(AO, tidx, ctx);
    }

    // 7) output projection -> f32 d_out
    gemm_proj<<<projBlocks, 256, 0, stream>>>(ctx, Wob, bo, d_out, 3, 1.0f);
}
